// Seq2SeqSNN_81097572483823
// MI455X (gfx1250) — compile-verified, matrix operand order checked by reading, not run on hardware
//
#include <hip/hip_runtime.h>

// ---------------------------------------------------------------------------
// Seq2SeqSNN forward for MI455X (gfx1250, wave32, WMMA).
//
// Decomposition: batch rows are fully independent through the recurrence, so
// 8 workgroups x 16 batch rows, each WG runs all SEQ_LEN*SPIKE_STEPS = 1792
// steps privately. 512 threads = 16 wave32s per WG; each wave owns a
// 32-column slice of HIDDEN (two 16x16 WMMA N-tiles). Membrane state stays in
// VGPRs in WMMA C-fragment layout; spikes are staged through LDS in WMMA
// A-fragment layout (stride 520 f16 -> 16B aligned, bank-conflict-free b128
// reads). Weights converted once to f16 into d_ws and streamed from L2.
// ---------------------------------------------------------------------------

#define SEQ_LEN     256
#define SPIKE_STEPS 7
#define BATCH       128
#define IN_DIM      96
#define HIDDEN      512
#define OUT_DIM     2

#define W0_ELEMS (HIDDEN * IN_DIM)    // 49152
#define W1_ELEMS (HIDDEN * HIDDEN)    // 262144
#define W2_ELEMS (HIDDEN * HIDDEN)    // 262144
#define WTOTAL   (W0_ELEMS + W1_ELEMS + W2_ELEMS)  // 573440

typedef __attribute__((ext_vector_type(16))) _Float16 v16h;
typedef __attribute__((ext_vector_type(8)))  _Float16 v8h;
typedef __attribute__((ext_vector_type(8)))  float    v8f;

// Load 16 f16 (32B) as two 16B chunks (LDS rows are only 16B-aligned).
__device__ __forceinline__ v16h ld_a(const _Float16* p) {
    const v8h* q = (const v8h*)p;
    v8h lo = q[0];
    v8h hi = q[1];
    return __builtin_shufflevector(lo, hi,
        0, 1, 2, 3, 4, 5, 6, 7, 8, 9, 10, 11, 12, 13, 14, 15);
}

// C = A(16xKD, from `src`, row stride ss) * B(KDx16 twice: hidden cols
// [hb,hb+16) and [hb+16,hb+32) of row-major W[H][KD]), accumulated in f32.
// A operand, contiguous k order used here (same as B): lane L holds M = L&15, K = (L>>4)*16 + e.
// B operand, same contiguous k order: lane L holds N = L&15, K = (L>>4)*16 + e, and
// B[k][n] = W[n][k], so per-lane data is 16 contiguous f16 of W's row n.
template <int KD>
__device__ __forceinline__ void mm2(const _Float16* __restrict__ W,
                                    const _Float16* src, int ss,
                                    int hb, int lr, int lh,
                                    v8f& c0, v8f& c1) {
    const _Float16* a0 = src + lr * ss + lh * 16;
    const _Float16* w0 = W + (hb + lr) * KD + lh * 16;
    const _Float16* w1 = w0 + 16 * KD;
#pragma unroll 4
    for (int k = 0; k < KD; k += 32) {
        v16h a  = ld_a(a0 + k);
        v16h b0 = ld_a(w0 + k);
        v16h b1 = ld_a(w1 + k);
        c0 = __builtin_amdgcn_wmma_f32_16x16x32_f16(
                 false, a, false, b0, (short)0, c0, false, false);
        c1 = __builtin_amdgcn_wmma_f32_16x16x32_f16(
                 false, a, false, b1, (short)0, c1, false, false);
    }
}

// LIF update on one C-fragment tile (layout: VGPR i, lane L ->
// M = i + 8*(L>>4), N = L&15), spike written to LDS in A-layout:
// zdst[M*ss + hb + N].
__device__ __forceinline__ void lif_store(v8f cur, v8f& mem, float beta,
                                          float thr, _Float16* zdst, int ss,
                                          int hb, int lr, int lh) {
    _Float16* base = zdst + hb + lr;
#pragma unroll
    for (int i = 0; i < 8; ++i) {
        float m    = mem[i];
        float keep = (m > thr) ? 0.0f : 1.0f;     // 1 - reset(mem_prev)
        m          = (beta * m + cur[i]) * keep;  // zero-reset mechanism
        mem[i]     = m;
        base[(i + lh * 8) * ss] = (m > thr) ? (_Float16)1.0f : (_Float16)0.0f;
    }
}

__global__ void convert_weights(const float* __restrict__ W0,
                                const float* __restrict__ W1,
                                const float* __restrict__ W2,
                                _Float16* __restrict__ ws) {
    int i = blockIdx.x * 256 + threadIdx.x;
    if (i < W0_ELEMS)
        ws[i] = (_Float16)W0[i];
    else if (i < W0_ELEMS + W1_ELEMS)
        ws[i] = (_Float16)W1[i - W0_ELEMS];
    else if (i < WTOTAL)
        ws[i] = (_Float16)W2[i - (W0_ELEMS + W1_ELEMS)];
}

__global__ __launch_bounds__(512) void snn_forward(
    const float* __restrict__ x,      // [256][128][96]
    const float* __restrict__ Wout,   // [2][512]
    const float* __restrict__ betas,  // [3]
    const float* __restrict__ thrs,   // [3]
    const _Float16* __restrict__ W0h, // [512][96]
    const _Float16* __restrict__ W1h, // [512][512]
    const _Float16* __restrict__ W2h, // [512][512]
    float* __restrict__ out)          // [256][128][2]
{
    constexpr int ZS = HIDDEN + 8;  // 520: 16B-aligned rows, conflict-free
    constexpr int AS = IN_DIM + 8;  // 104

    __shared__ _Float16 abuf[16 * AS];           // x tile (f16, A-layout)
    __shared__ _Float16 zb0[16 * ZS];            // layer-0 spikes
    __shared__ _Float16 zb1[16 * ZS];            // layer-1 spikes
    __shared__ _Float16 zb2[16 * ZS];            // layer-2 spikes
    __shared__ float    woutl[OUT_DIM * HIDDEN]; // Wout, f32
    __shared__ float    part[16 * 32];           // per-wave output partials

    const int tid = threadIdx.x;
    const int wv  = tid >> 5;    // wave 0..15
    const int ln  = tid & 31;    // lane
    const int lr  = ln & 15;     // fragment row/col select
    const int lh  = ln >> 4;     // fragment half select
    const int br0 = blockIdx.x * 16;  // first batch row of this WG
    const int hb  = wv * 32;          // this wave's hidden-column base

    for (int i = tid; i < OUT_DIM * HIDDEN; i += 512) woutl[i] = Wout[i];

    const float be0 = betas[0], be1 = betas[1], be2 = betas[2];
    const float th0 = thrs[0],  th1 = thrs[1],  th2 = thrs[2];

    v8f mems[3][2] = {};      // membrane state, C-fragment layout
    float m_out = 0.0f;       // integrator value for this lane's (row,out)
    const int r2 = ln >> 1;   // output-stage row    (0..15)
    const int oo = ln & 1;    // output-stage column (0..1)

    for (int s = 0; s < SEQ_LEN; ++s) {
        // Stage x[s] rows br0..br0+15 into LDS as f16 A-layout (shared by
        // all 7 inner spike steps of this sequence step).
        for (int i = tid; i < 16 * IN_DIM; i += 512) {
            int r = i / IN_DIM;
            int c = i - r * IN_DIM;
            abuf[r * AS + c] =
                (_Float16)x[((s * BATCH) + br0 + r) * IN_DIM + c];
        }
        __syncthreads();

        float accO = 0.0f;  // sum of integrator values over spike steps

        for (int k = 0; k < SPIKE_STEPS; ++k) {
            // ---- Layer 0: cur = x @ W0^T (K = 96) ----
            {
                v8f c0 = {}, c1 = {};
                mm2<IN_DIM>(W0h, abuf, AS, hb, lr, lh, c0, c1);
                lif_store(c0, mems[0][0], be0, th0, zb0, ZS, hb,      lr, lh);
                lif_store(c1, mems[0][1], be0, th0, zb0, ZS, hb + 16, lr, lh);
            }
            __syncthreads();

            // ---- Layer 1: cur = z0 @ W1^T (K = 512) ----
            {
                v8f c0 = {}, c1 = {};
                mm2<HIDDEN>(W1h, zb0, ZS, hb, lr, lh, c0, c1);
                lif_store(c0, mems[1][0], be1, th1, zb1, ZS, hb,      lr, lh);
                lif_store(c1, mems[1][1], be1, th1, zb1, ZS, hb + 16, lr, lh);
            }
            __syncthreads();

            // ---- Layer 2: cur = z1 @ W2^T (K = 512) ----
            {
                v8f c0 = {}, c1 = {};
                mm2<HIDDEN>(W2h, zb1, ZS, hb, lr, lh, c0, c1);
                lif_store(c0, mems[2][0], be2, th2, zb2, ZS, hb,      lr, lh);
                lif_store(c1, mems[2][1], be2, th2, zb2, ZS, hb + 16, lr, lh);
            }
            __syncthreads();

            // ---- Output integrator: mem_out += z2 @ Wout^T ----
            // Each wave: partial dot over its 32 hidden columns for the
            // (row, out) pair owned by each lane.
            {
                const _Float16* zrow = &zb2[r2 * ZS + hb];
                const float*    wrow = &woutl[oo * HIDDEN + hb];
                float p = 0.0f;
#pragma unroll 8
                for (int c = 0; c < 32; ++c)
                    p += (float)zrow[c] * wrow[c];
                part[wv * 32 + ln] = p;
            }
            __syncthreads();

            // Fixed-order reduction (deterministic), redundantly in all
            // waves so the integrator state is replicated per lane.
            {
                float inc = 0.0f;
#pragma unroll
                for (int w = 0; w < 16; ++w) inc += part[w * 32 + ln];
                m_out += inc;
                accO  += m_out;  // mean-decoder accumulator
            }
        }

        if (tid < 32)
            out[((s * BATCH) + br0 + r2) * OUT_DIM + oo] =
                accO * (1.0f / (float)SPIKE_STEPS);
    }
}

extern "C" void kernel_launch(void* const* d_in, const int* in_sizes, int n_in,
                              void* d_out, int out_size, void* d_ws,
                              size_t ws_size, hipStream_t stream) {
    const float* x     = (const float*)d_in[0];
    const float* W0    = (const float*)d_in[1];
    const float* W1    = (const float*)d_in[2];
    const float* W2    = (const float*)d_in[3];
    const float* Wout  = (const float*)d_in[4];
    const float* betas = (const float*)d_in[5];
    const float* thrs  = (const float*)d_in[6];

    // d_ws layout: f16 weights [W0 | W1 | W2], 1,146,880 bytes total.
    _Float16* wsH = (_Float16*)d_ws;
    const _Float16* W0h = wsH;
    const _Float16* W1h = wsH + W0_ELEMS;
    const _Float16* W2h = wsH + W0_ELEMS + W1_ELEMS;

    convert_weights<<<(WTOTAL + 255) / 256, 256, 0, stream>>>(W0, W1, W2, wsH);
    snn_forward<<<BATCH / 16, 512, 0, stream>>>(
        x, Wout, betas, thrs, W0h, W1h, W2h, (float*)d_out);
}
